// MPNN_87110526697921
// MI455X (gfx1250) — matrix operand order repaired and checked by reading, not compiled, not run on hardware
//
#include <hip/hip_runtime.h>
#include <hip/hip_bf16.h>
#include <math.h>

// ---------------------------------------------------------------------------
// MI455X (gfx1250) MPNN pipeline.
// GEMMs: v_wmma_f32_16x16x32_bf16 (wave32, f32 accum), double-buffered LDS,
//        A-tiles staged via GLOBAL_LOAD_ASYNC_TO_LDS_B128 (ASYNCcnt) when
//        available, weight prefetch via global_prefetch_b8.
// Scatter/segment sums: native global_atomic_add_f32 (L2-resident working set).
// ---------------------------------------------------------------------------

typedef __attribute__((ext_vector_type(16))) __bf16 v16bf;
typedef __attribute__((ext_vector_type(8)))  float  v8f;

#if __has_builtin(__builtin_amdgcn_global_load_async_to_lds_b128)
#define HAVE_ASYNC_LDS 1
typedef int v4i_ __attribute__((__vector_size__(16)));
typedef __attribute__((address_space(1))) v4i_* gp4_t;   // global int4*
typedef __attribute__((address_space(3))) v4i_* lp3_t;   // LDS int4*
#endif

#define H       256
#define HID1P   320       // 263 padded up to a multiple of 64 (N-tile) & 32 (K-tile)
#define KIN     32        // 10/12 input features padded to one K step
#define NCELLS  35000
#define NTRK    5000
#define NPRE    40000
#define NTOPO   2000
#define NNODES  7000
#define EPP     640000
#define KCAT    768       // agg(256) | pre_h(256) | g(256)
#define LDCOUT  263
#define EPSN    1e-8f

__device__ __forceinline__ __bf16 f2bf(float f) {
  unsigned u = __builtin_bit_cast(unsigned, f);
  unsigned r = u + 0x7FFFu + ((u >> 16) & 1u);     // round-to-nearest-even
  unsigned short s = (unsigned short)(r >> 16);
  return __builtin_bit_cast(__bf16, s);
}

__device__ __forceinline__ void wait_async0() {
#if defined(HAVE_ASYNC_LDS)
#if __has_builtin(__builtin_amdgcn_s_wait_asynccnt)
  __builtin_amdgcn_s_wait_asynccnt(0);
#else
  asm volatile("s_wait_asynccnt 0x0" ::: "memory");
#endif
#endif
}

// ---------------------------------------------------------------------------
// WMMA GEMM:  out = act(A[MxK]bf16 @ W[KxN]bf16 + bias)
// Block: 256 threads = 8 waves; tile 128(M) x 64(N), K-step 32, 2 LDS stages.
// Waves: 4(M) x 2(N); each wave owns 32x32 = 2x2 WMMA 16x16 tiles.
// Fragment layouts per CDNA5 ISA 7.12.2 (16-bit A 16x32, B 32x16, f32 C/D).
// ---------------------------------------------------------------------------
#define TM 128
#define TN 64
#define TK 32
#define LDA 48    // LDS row stride (elems): 96B -> b128-aligned & bank-staggered
#define LDB 48

template<bool RELU, bool OUTF, bool OUTB>
__global__ __launch_bounds__(256)
void gemm_bf16(const __bf16* __restrict__ A, const __bf16* __restrict__ W,
               const float* __restrict__ bias,
               float* __restrict__ outF, int ldc,
               __bf16* __restrict__ outB,
               int M, int N, int K)
{
  __shared__ __bf16 As[2][TM * LDA];
  __shared__ __bf16 Bs[2][TN * LDB];   // stored transposed: [n][k]

  const int tid  = threadIdx.x;
  const int lane = tid & 31;
  const int wave = tid >> 5;
  const int wm   = wave & 3;
  const int wn   = wave >> 2;
  const int bm   = blockIdx.x * TM;
  const int bn   = blockIdx.y * TN;
  const bool fullM = (bm + TM <= M);   // uniform per block

  v8f acc[2][2] = {};

  const int arow0 = tid >> 2;          // A: 4 threads/row, 8 bf16 each, 2 passes
  const int acol  = (tid & 3) * 8;
  const int bkrow = tid >> 3;          // B: 8 threads/row along N, 32 K-rows
  const int bncol = (tid & 7) * 8;

  const int mloc = lane & 15;
  const int hi   = (lane >> 4) & 1;

  // ---- tile loaders -------------------------------------------------------
  auto loadA = [&](int k0, int st) {
#if defined(HAVE_ASYNC_LDS)
    if (fullM) {
      // CDNA5 async DMA: global -> LDS, no VGPR staging, tracked by ASYNCcnt
#pragma unroll
      for (int p = 0; p < 2; ++p) {
        int r = arow0 + p * 64;
        const __bf16* src = A + (size_t)(bm + r) * K + k0 + acol;
        __bf16* dst = &As[st][r * LDA + acol];
        __builtin_amdgcn_global_load_async_to_lds_b128(
            (gp4_t)(unsigned long long)src,
            (lp3_t)(unsigned)(unsigned long long)dst, 0, 0);
      }
      return;
    }
#endif
#pragma unroll
    for (int p = 0; p < 2; ++p) {       // guarded sync path (edge M blocks)
      int r  = arow0 + p * 64;
      int gr = bm + r;
      uint4 v = make_uint4(0u, 0u, 0u, 0u);
      if (gr < M) v = *(const uint4*)(A + (size_t)gr * K + k0 + acol);
      *(uint4*)(&As[st][r * LDA + acol]) = v;
    }
  };

  auto loadB = [&](int k0, int st) {    // transpose store (N mult of 64, K of 32)
    uint4 v = *(const uint4*)(W + (size_t)(k0 + bkrow) * N + bn + bncol);
    const __bf16* hh = (const __bf16*)&v;
#pragma unroll
    for (int j = 0; j < 8; ++j) Bs[st][(bncol + j) * LDB + bkrow] = hh[j];
  };

  // ---- software pipeline --------------------------------------------------
  loadA(0, 0);
  loadB(0, 0);
  wait_async0();
  __syncthreads();

  const int nk = K / TK;
  for (int t = 0; t < nk; ++t) {
    const int st = t & 1;
    if (t + 1 < nk) {
      loadA((t + 1) * TK, st ^ 1);      // overlap with WMMA below
      loadB((t + 1) * TK, st ^ 1);
      if (t + 2 < nk)                   // prefetch weight tile t+2 into L2/L0
        __builtin_prefetch(W + (size_t)(t + 2) * TK * N + bn, 0, 3);
    }

    v16bf af[2], bfr[2];
#pragma unroll
    for (int q = 0; q < 2; ++q) {
      const int rowA = wm * 32 + q * 16 + mloc;
#pragma unroll
      for (int j = 0; j < 8; ++j) {     // A 16x32: VGPR j holds K pair
        int k = hi * 8 + (j & 3) * 2 + (j >> 2) * 16;
        af[q][2 * j]     = As[st][rowA * LDA + k];
        af[q][2 * j + 1] = As[st][rowA * LDA + k + 1];
      }
      const int rowB = wn * 32 + q * 16 + mloc;
#pragma unroll
      for (int j = 0; j < 8; ++j) {     // B 32x16: same K order as A (lanes 0-15 K=0..7,16..23; 16-31 K=8..15,24..31)
        int k = hi * 8 + (j & 3) * 2 + (j >> 2) * 16;
        bfr[q][2 * j]     = Bs[st][rowB * LDB + k];
        bfr[q][2 * j + 1] = Bs[st][rowB * LDB + k + 1];
      }
    }
#pragma unroll
    for (int i = 0; i < 2; ++i)
#pragma unroll
      for (int j = 0; j < 2; ++j)
        acc[i][j] = __builtin_amdgcn_wmma_f32_16x16x32_bf16(
            false, af[i], false, bfr[j], (short)0, acc[i][j], false, false);

    wait_async0();                      // next-tile DMA arrival
    __syncthreads();
  }

  // ---- epilogue: C/D layout -> VGPR r: M = r + 8*hi, N = lane&15
#pragma unroll
  for (int i = 0; i < 2; ++i) {
#pragma unroll
    for (int j = 0; j < 2; ++j) {
      const int col   = bn + wn * 32 + j * 16 + mloc;
      const float bv  = bias ? bias[col] : 0.f;
      const int rbase = bm + wm * 32 + i * 16 + hi * 8;
#pragma unroll
      for (int r = 0; r < 8; ++r) {
        int row = rbase + r;
        if (row < M) {
          float v = acc[i][j][r] + bv;
          if (RELU) v = fmaxf(v, 0.f);
          if (OUTF) outF[(size_t)row * ldc + col] = v;
          if (OUTB) outB[(size_t)row * N + col]   = f2bf(v);
        }
      }
    }
  }
}

// ---------------------------------------------------------------------------
// Utility / scatter kernels
// ---------------------------------------------------------------------------
__global__ void k_zero(float* p, int n) {
  int i = blockIdx.x * blockDim.x + threadIdx.x;
  if (i < n) p[i] = 0.f;
}

__global__ void k_padcvt(const float* __restrict__ src, int R, int C,
                         __bf16* __restrict__ dst, int DR, int DC) {
  int i = blockIdx.x * blockDim.x + threadIdx.x;
  if (i >= DR * DC) return;
  int r = i / DC, c = i - r * DC;
  float v = (r < R && c < C) ? src[r * C + c] : 0.f;
  dst[i] = f2bf(v);
}

__global__ void k_padbias(const float* __restrict__ src, int C,
                          float* __restrict__ dst, int DC) {
  int i = blockIdx.x * blockDim.x + threadIdx.x;
  if (i < DC) dst[i] = (i < C) ? src[i] : 0.f;
}

__global__ __launch_bounds__(256)
void k_scatter_edges(const float* __restrict__ pre, const int* __restrict__ src,
                     const int* __restrict__ dst, float* __restrict__ agg) {
  int t = blockIdx.x * 256 + threadIdx.x;   // EPP*64 threads, 4 floats each
  int e = t >> 6, c = (t & 63) << 2;
  if (e >= EPP) return;
  int s = src[e], d = dst[e];
  const float4 v = *(const float4*)(pre + (size_t)s * H + c);
  float* o = agg + (size_t)d * H + c;
  unsafeAtomicAdd(o + 0, v.x); unsafeAtomicAdd(o + 1, v.y);
  unsafeAtomicAdd(o + 2, v.z); unsafeAtomicAdd(o + 3, v.w);
}

__global__ void k_pack_inp(const float* __restrict__ agg, const float* __restrict__ pre,
                           const float* __restrict__ g, __bf16* __restrict__ inp) {
  int t = blockIdx.x * blockDim.x + threadIdx.x;   // NPRE*192 threads, 4 elems each
  if (t >= NPRE * 192) return;
  int row = t / 192, c4 = (t - row * 192) * 4;
  float v[4];
#pragma unroll
  for (int j = 0; j < 4; ++j) {
    int c = c4 + j;
    v[j] = (c < 256) ? agg[(size_t)row * H + c]
         : (c < 512) ? pre[(size_t)row * H + (c - 256)]
                     : g[c - 512];
  }
  __bf16* o = inp + (size_t)row * KCAT + c4;
#pragma unroll
  for (int j = 0; j < 4; ++j) o[j] = f2bf(v[j]);
}

__global__ __launch_bounds__(256)
void k_colsum(const float* __restrict__ pre, float* __restrict__ gacc, int rows) {
  int c = threadIdx.x;
  int r0 = blockIdx.x * 256;
  int rend = min(r0 + 256, rows);
  float s = 0.f;
  for (int r = r0; r < rend; ++r) s += pre[(size_t)r * H + c];
  unsafeAtomicAdd(gacc + c, s);
}

__global__ __launch_bounds__(256)
void k_rownorm(const float* __restrict__ raw, float* __restrict__ pre,
               float* __restrict__ gacc) {
  __shared__ float red[8];
  int row = blockIdx.x, c = threadIdx.x;
  float v = raw[(size_t)row * H + c];
  float ss = v * v;
#pragma unroll
  for (int off = 16; off > 0; off >>= 1) ss += __shfl_xor(ss, off, 32);
  if ((c & 31) == 0) red[c >> 5] = ss;
  __syncthreads();
  float tot = 0.f;
#pragma unroll
  for (int i = 0; i < 8; ++i) tot += red[i];
  float nv = v / (sqrtf(tot) + EPSN);
  pre[(size_t)row * H + c] = nv;
  unsafeAtomicAdd(gacc + c, nv);
}

__global__ void k_gnorm(const float* __restrict__ gacc, float* __restrict__ g) {
  __shared__ float red[8];
  int c = threadIdx.x;
  float v = gacc[c];
  float ss = v * v;
#pragma unroll
  for (int off = 16; off > 0; off >>= 1) ss += __shfl_xor(ss, off, 32);
  if ((c & 31) == 0) red[c >> 5] = ss;
  __syncthreads();
  float tot = 0.f;
#pragma unroll
  for (int i = 0; i < 8; ++i) tot += red[i];
  g[c] = v / (sqrtf(tot) + EPSN);
}

__global__ __launch_bounds__(256)
void k_scatter_topo(const float* __restrict__ pre, const int* __restrict__ ptdst,
                    float* __restrict__ out) {
  int t = blockIdx.x * 256 + threadIdx.x;   // NPRE*64 threads
  int r = t >> 6, c = (t & 63) << 2;
  if (r >= NPRE) return;
  int d = ptdst[r];
  const float4 v = *(const float4*)(pre + (size_t)r * H + c);
  float* o = out + (size_t)d * LDCOUT + c;
  unsafeAtomicAdd(o + 0, v.x); unsafeAtomicAdd(o + 1, v.y);
  unsafeAtomicAdd(o + 2, v.z); unsafeAtomicAdd(o + 3, v.w);
}

__global__ void k_copy_tracks(const float* __restrict__ ht, float* __restrict__ out) {
  int i = blockIdx.x * blockDim.x + threadIdx.x;
  if (i >= NTRK * H) return;
  int r = i / H, c = i - r * H;
  out[(size_t)(NTOPO + r) * LDCOUT + c] = ht[i];
}

__global__ void k_cell_phys(const float* __restrict__ ceta, const float* __restrict__ cphi,
                            const float* __restrict__ ce, const float* __restrict__ clay,
                            const int* __restrict__ ctdst,
                            float* __restrict__ E, float* __restrict__ weta,
                            float* __restrict__ wphi, float* __restrict__ wlay) {
  int i = blockIdx.x * blockDim.x + threadIdx.x;
  if (i >= NCELLS) return;
  float eta = ceta[i] * 1.5f;
  float phi = cphi[i] * 1.8f;
  float en  = expf(ce[i] * 1.0f + 2.0f);
  int d = ctdst[i];
  unsafeAtomicAdd(E + d, en);
  unsafeAtomicAdd(weta + d, eta * en);
  unsafeAtomicAdd(wphi + d, phi * en);
  unsafeAtomicAdd(wlay + d, clay[i] * en);
}

__global__ void k_node_feats(const float* __restrict__ E, const float* __restrict__ weta,
                             const float* __restrict__ wphi, const float* __restrict__ wlay,
                             const float* __restrict__ teta, const float* __restrict__ tphi,
                             const float* __restrict__ te,   const float* __restrict__ tlay,
                             const float* __restrict__ isTrack, const float* __restrict__ isMuon,
                             const float* __restrict__ tpt, float* __restrict__ out) {
  int i = blockIdx.x * blockDim.x + threadIdx.x;
  if (i >= NNODES) return;
  float eta, phi, en, lay;
  if (i < NTOPO) {
    float e = E[i];
    eta = (weta[i] / e) / 1.5f;
    phi = (wphi[i] / e) / 1.8f;
    lay = wlay[i] / e;
    en  = (logf(e) - 3.0f) / 1.5f;
  } else {
    int j = i - NTOPO;
    eta = teta[j]; phi = tphi[j]; en = te[j]; lay = tlay[j];
  }
  float* o = out + (size_t)i * LDCOUT;
  o[256] = eta; o[257] = phi; o[258] = en; o[259] = lay;
  o[260] = isTrack[i]; o[261] = isMuon[i]; o[262] = tpt[i];
}

// ---------------------------------------------------------------------------
// Host orchestration
// ---------------------------------------------------------------------------
static inline size_t alignup(size_t x) { return (x + 255) & ~(size_t)255; }

extern "C" void kernel_launch(void* const* d_in, const int* in_sizes, int n_in,
                              void* d_out, int out_size, void* d_ws, size_t ws_size,
                              hipStream_t stream) {
  (void)in_sizes; (void)n_in; (void)out_size; (void)ws_size;

  // --- input pointers (setup_inputs dict order; params flattened as a jax
  //     pytree: sorted keys -> blocks, cell_init, track_init; "W" before "b")
  const float* cell_feats = (const float*)d_in[0];
  const float* track_feats= (const float*)d_in[1];
  const float* cell_eta   = (const float*)d_in[2];
  const float* cell_phi   = (const float*)d_in[3];
  const float* cell_e     = (const float*)d_in[4];
  const float* cell_layer = (const float*)d_in[5];
  const float* track_eta  = (const float*)d_in[6];
  const float* track_phi  = (const float*)d_in[7];
  const float* track_e    = (const float*)d_in[8];
  const float* track_layer= (const float*)d_in[9];
  const float* isTrack    = (const float*)d_in[10];
  const float* isMuon     = (const float*)d_in[11];
  const float* track_pt   = (const float*)d_in[12];
  const float* blkW[2][3] = {{(const float*)d_in[13], (const float*)d_in[15], (const float*)d_in[17]},
                             {(const float*)d_in[19], (const float*)d_in[21], (const float*)d_in[23]}};
  const float* blkB[2][3] = {{(const float*)d_in[14], (const float*)d_in[16], (const float*)d_in[18]},
                             {(const float*)d_in[20], (const float*)d_in[22], (const float*)d_in[24]}};
  const float* cW1 = (const float*)d_in[25];  const float* cB1 = (const float*)d_in[26];
  const float* cW2 = (const float*)d_in[27];  const float* cB2 = (const float*)d_in[28];
  const float* tW1 = (const float*)d_in[29];  const float* tB1 = (const float*)d_in[30];
  const float* tW2 = (const float*)d_in[31];  const float* tB2 = (const float*)d_in[32];
  const int* pp_src = (const int*)d_in[33];
  const int* pp_dst = (const int*)d_in[34];
  const int* pt_dst = (const int*)d_in[36];   // pt_src == arange
  const int* ct_dst = (const int*)d_in[38];   // ct_src == arange
  float* out = (float*)d_out;

  // --- workspace carve-out
  char* wsb = (char*)d_ws;
  size_t off = 0;
  auto carve = [&](size_t bytes) -> void* { void* p = wsb + off; off = alignup(off + bytes); return p; };

  float*  pre_h  = (float*) carve((size_t)NPRE * H * 4);
  float*  agg    = (float*) carve((size_t)NPRE * H * 4);   // also "raw" GEMM3 out
  __bf16* inp    = (__bf16*)carve((size_t)NPRE * KCAT * 2);
  __bf16* hid1   = (__bf16*)carve((size_t)NPRE * H * 2);
  __bf16* hid2   = (__bf16*)carve((size_t)NPRE * H * 2);
  float*  htrk   = (float*) carve((size_t)NTRK * H * 4);
  float*  g      = (float*) carve(H * 4);
  float*  gacc   = (float*) carve(H * 4);
  float*  Ebuf   = (float*) carve((size_t)4 * NTOPO * 4);  // E | weta | wphi | wlay
  float*  wetaB  = Ebuf + NTOPO;
  float*  wphiB  = Ebuf + 2 * NTOPO;
  float*  wlayB  = Ebuf + 3 * NTOPO;
  __bf16* Wc1  = (__bf16*)carve((size_t)KIN * HID1P * 2);
  float*  bc1  = (float*) carve(HID1P * 4);
  __bf16* Wc2  = (__bf16*)carve((size_t)HID1P * H * 2);
  __bf16* Wt1  = (__bf16*)carve((size_t)KIN * HID1P * 2);
  float*  bt1  = (float*) carve(HID1P * 4);
  __bf16* Wt2  = (__bf16*)carve((size_t)HID1P * H * 2);
  __bf16* WB[2][3];
  WB[0][0] = (__bf16*)carve((size_t)KCAT * H * 2);
  WB[0][1] = (__bf16*)carve((size_t)H * H * 2);
  WB[0][2] = (__bf16*)carve((size_t)H * H * 2);
  WB[1][0] = (__bf16*)carve((size_t)KCAT * H * 2);
  WB[1][1] = (__bf16*)carve((size_t)H * H * 2);
  WB[1][2] = (__bf16*)carve((size_t)H * H * 2);

  // init-phase buffers aliased inside `inp` (free until first pack)
  char* ib = (char*)inp;
  size_t ioff = 0;
  auto icarve = [&](size_t bytes) -> void* { void* p = ib + ioff; ioff = alignup(ioff + bytes); return p; };
  __bf16* cellsA = (__bf16*)icarve((size_t)NCELLS * KIN * 2);
  __bf16* trkA   = (__bf16*)icarve((size_t)NTRK * KIN * 2);
  __bf16* cellH1 = (__bf16*)icarve((size_t)NCELLS * HID1P * 2);
  __bf16* trkH1  = (__bf16*)icarve((size_t)NTRK * HID1P * 2);

  const dim3 B256(256);
  auto g1d = [](long n) { return dim3((unsigned)((n + 255) / 256)); };

  // ---- phase 0: convert weights to bf16 (zero-padded), pack A inputs
  k_padcvt<<<g1d((long)KIN * HID1P), B256, 0, stream>>>(cW1, 10, 263, Wc1, KIN, HID1P);
  k_padbias<<<g1d(HID1P), B256, 0, stream>>>(cB1, 263, bc1, HID1P);
  k_padcvt<<<g1d((long)HID1P * H), B256, 0, stream>>>(cW2, 263, 256, Wc2, HID1P, H);
  k_padcvt<<<g1d((long)KIN * HID1P), B256, 0, stream>>>(tW1, 12, 263, Wt1, KIN, HID1P);
  k_padbias<<<g1d(HID1P), B256, 0, stream>>>(tB1, 263, bt1, HID1P);
  k_padcvt<<<g1d((long)HID1P * H), B256, 0, stream>>>(tW2, 263, 256, Wt2, HID1P, H);
  for (int b = 0; b < 2; ++b) {
    k_padcvt<<<g1d((long)KCAT * H), B256, 0, stream>>>(blkW[b][0], KCAT, H, WB[b][0], KCAT, H);
    k_padcvt<<<g1d((long)H * H), B256, 0, stream>>>(blkW[b][1], H, H, WB[b][1], H, H);
    k_padcvt<<<g1d((long)H * H), B256, 0, stream>>>(blkW[b][2], H, H, WB[b][2], H, H);
  }
  k_padcvt<<<g1d((long)NCELLS * KIN), B256, 0, stream>>>(cell_feats, NCELLS, 10, cellsA, NCELLS, KIN);
  k_padcvt<<<g1d((long)NTRK * KIN), B256, 0, stream>>>(track_feats, NTRK, 12, trkA, NTRK, KIN);

  // ---- phase 1: init MLPs -> pre_h
  gemm_bf16<true, false, true><<<dim3((NCELLS + TM - 1) / TM, HID1P / TN), B256, 0, stream>>>(
      cellsA, Wc1, bc1, nullptr, 0, cellH1, NCELLS, HID1P, KIN);
  gemm_bf16<false, true, false><<<dim3((NCELLS + TM - 1) / TM, H / TN), B256, 0, stream>>>(
      cellH1, Wc2, cB2, pre_h, H, nullptr, NCELLS, H, HID1P);
  gemm_bf16<true, false, true><<<dim3((NTRK + TM - 1) / TM, HID1P / TN), B256, 0, stream>>>(
      trkA, Wt1, bt1, nullptr, 0, trkH1, NTRK, HID1P, KIN);
  gemm_bf16<false, true, false><<<dim3((NTRK + TM - 1) / TM, H / TN), B256, 0, stream>>>(
      trkH1, Wt2, tB2, pre_h + (size_t)NCELLS * H, H, nullptr, NTRK, H, HID1P);
  (void)hipMemcpyAsync(htrk, pre_h + (size_t)NCELLS * H, (size_t)NTRK * H * 4,
                       hipMemcpyDeviceToDevice, stream);

  k_zero<<<1, 256, 0, stream>>>(gacc, H);
  k_colsum<<<dim3((NPRE + 255) / 256), B256, 0, stream>>>(pre_h, gacc, NPRE);
  k_gnorm<<<1, 256, 0, stream>>>(gacc, g);

  // ---- phase 2: 4 GN iterations (2 blocks x 2 iters)
  for (int it = 0; it < 4; ++it) {
    const int bi = it >> 1;
    k_zero<<<g1d((long)NPRE * H), B256, 0, stream>>>(agg, NPRE * H);
    k_scatter_edges<<<dim3(EPP * 64 / 256), B256, 0, stream>>>(pre_h, pp_src, pp_dst, agg);
    k_pack_inp<<<g1d((long)NPRE * 192), B256, 0, stream>>>(agg, pre_h, g, inp);
    gemm_bf16<true, false, true><<<dim3((NPRE + TM - 1) / TM, H / TN), B256, 0, stream>>>(
        inp, WB[bi][0], blkB[bi][0], nullptr, 0, hid1, NPRE, H, KCAT);
    gemm_bf16<true, false, true><<<dim3((NPRE + TM - 1) / TM, H / TN), B256, 0, stream>>>(
        hid1, WB[bi][1], blkB[bi][1], nullptr, 0, hid2, NPRE, H, H);
    gemm_bf16<false, true, false><<<dim3((NPRE + TM - 1) / TM, H / TN), B256, 0, stream>>>(
        hid2, WB[bi][2], blkB[bi][2], agg, H, nullptr, NPRE, H, H);  // raw -> agg buf
    k_zero<<<1, 256, 0, stream>>>(gacc, H);
    k_rownorm<<<dim3(NPRE), B256, 0, stream>>>(agg, pre_h, gacc);
    k_gnorm<<<1, 256, 0, stream>>>(gacc, g);
  }

  // ---- phase 3: outputs
  k_zero<<<g1d((long)NNODES * LDCOUT), B256, 0, stream>>>(out, NNODES * LDCOUT);
  k_scatter_topo<<<dim3(NPRE * 64 / 256), B256, 0, stream>>>(pre_h, pt_dst, out);
  k_copy_tracks<<<g1d((long)NTRK * H), B256, 0, stream>>>(htrk, out);
  k_zero<<<g1d(4L * NTOPO), B256, 0, stream>>>(Ebuf, 4 * NTOPO);
  k_cell_phys<<<g1d(NCELLS), B256, 0, stream>>>(cell_eta, cell_phi, cell_e, cell_layer,
                                                ct_dst, Ebuf, wetaB, wphiB, wlayB);
  k_node_feats<<<g1d(NNODES), B256, 0, stream>>>(Ebuf, wetaB, wphiB, wlayB,
                                                 track_eta, track_phi, track_e, track_layer,
                                                 isTrack, isMuon, track_pt, out);
}
